// MorphModel_36670430773842
// MI455X (gfx1250) — matrix operand order repaired and checked by reading, not compiled, not run on hardware
//
#include <hip/hip_runtime.h>
#include <hip/hip_bf16.h>
#include <stdint.h>

// ---------------- problem constants ----------------
constexpr int B = 1024, I = 512, H = 2048, O = 512, E = 8;

// ---------------- GEMM tiling ----------------
constexpr int BM = 128;   // block tile M
constexpr int BN = 256;   // block tile N
constexpr int BK = 32;    // k step
constexpr int LDA = 40;   // padded LDS row stride (halves): 80B, 16B-aligned, bank-spread
constexpr int LDB = 40;

typedef _Float16 half2_t  __attribute__((ext_vector_type(2)));
typedef _Float16 half4_t  __attribute__((ext_vector_type(4)));
typedef _Float16 half8_t  __attribute__((ext_vector_type(8)));
typedef _Float16 half16_t __attribute__((ext_vector_type(16)));
typedef float    float8_t __attribute__((ext_vector_type(8)));
typedef unsigned int uint32x4 __attribute__((ext_vector_type(4)));
typedef unsigned int uint32x8 __attribute__((ext_vector_type(8)));

// ---------------- CDNA5 async data movers ----------------
// Per-lane gather: LDS[lds_addr] <= MEM[src], 16 bytes, tracked by ASYNCcnt.
__device__ __forceinline__ void async_b128(uint32_t lds_addr, const void* src) {
    asm volatile("global_load_async_to_lds_b128 %0, %1, off"
                 :: "v"(lds_addr), "v"((unsigned long long)(uintptr_t)src)
                 : "memory");
}
__device__ __forceinline__ void wait_async0() {
    asm volatile("s_wait_asynccnt 0x0" ::: "memory");
}

// TDM 2D tile load: tile_d1 rows x tile_d0 elements (f16), row stride stride0 (elems),
// rows >= tensor_d1 read as zero.  LDS rows padded +4 DWORDs every 16 DWORDs -> stride 40 halves.
__device__ __forceinline__ void tdm_load_2d(uint32_t lds_addr, const void* gptr,
                                            uint32_t tensor_d0, uint32_t tensor_d1,
                                            uint32_t tile_d0, uint32_t tile_d1,
                                            uint64_t stride0) {
    const uint64_t ga = (uint64_t)(uintptr_t)gptr;
    uint32x4 g0;
    g0[0] = 1u;                                   // count=1, user descriptor
    g0[1] = lds_addr;                             // LDS byte address
    g0[2] = (uint32_t)ga;                         // global_addr[31:0]
    g0[3] = (uint32_t)(ga >> 32) | (2u << 30);    // global_addr[56:32] | type=2
    uint32x8 g1;
    g1[0] = (1u << 16)                            // data_size = 1 -> 2 bytes
          | (1u << 20)                            // pad_enable
          | (3u << 22)                            // pad_interval: 16 DWORDs
          | (3u << 25);                           // pad_amount:   4 DWORDs
    g1[1] = (tensor_d0 & 0xFFFFu) << 16;          // tensor_dim0[15:0] @bit48
    g1[2] = (tensor_d0 >> 16) | ((tensor_d1 & 0xFFFFu) << 16);
    g1[3] = (tensor_d1 >> 16) | (tile_d0 << 16);  // tile_dim0 @bit112
    g1[4] = tile_d1 & 0xFFFFu;                    // tile_dim1, tile_dim2=0
    g1[5] = (uint32_t)stride0;                    // dim0_stride[31:0]
    g1[6] = (uint32_t)(stride0 >> 32) & 0xFFFFu;  // dim0_stride[47:32], dim1_stride lo=0
    g1[7] = 0u;
    asm volatile("tensor_load_to_lds %0, %1" :: "s"(g0), "s"(g1) : "memory");
}

// ---- WMMA fragment loaders (ISA 7.12.2 layouts, wave32) ----
__device__ __forceinline__ half16_t frag_a(const _Float16* As, int mbase, int lane) {
    int m  = mbase + (lane & 15);
    int kb = (lane >> 4) << 3;          // 0 or 8
    const _Float16* p = As + m * LDA + kb;
    half8_t lo = *(const half8_t*)p;          // K kb..kb+7
    half8_t hi = *(const half8_t*)(p + 16);   // K kb+16..kb+23
    return __builtin_shufflevector(lo, hi, 0,1,2,3,4,5,6,7,8,9,10,11,12,13,14,15);
}
__device__ __forceinline__ half16_t frag_b(const _Float16* Bs, int nbase, int lane) {
    int n  = nbase + (lane & 15);
    int kb = (lane >> 4) << 3;          // 0 or 8
    const _Float16* p = Bs + n * LDB + kb;
    half8_t lo = *(const half8_t*)p;
    half8_t hi = *(const half8_t*)(p + 16);
    return __builtin_shufflevector(lo, hi, 0,1,2,3,4,5,6,7,8,9,10,11,12,13,14,15);
}

// ---------------- precision prep: x -> f16 ----------------
__global__ void __launch_bounds__(256) cvt_x_kernel(const float* __restrict__ x,
                                                    _Float16* __restrict__ xh) {
    int i = blockIdx.x * 256 + threadIdx.x;       // float4 granules
    float4 v = ((const float4*)x)[i];
    half4_t h = { (_Float16)v.x, (_Float16)v.y, (_Float16)v.z, (_Float16)v.w };
    ((half4_t*)xh)[i] = h;
}

// ---------------- W [E][K][N] f32 -> Wt [E][N][K] f16 (tiled transpose) ----------------
__global__ void __launch_bounds__(256) transpose_cvt_kernel(const float* __restrict__ W,
                                                            _Float16* __restrict__ Wt,
                                                            int K, int N) {
    __shared__ _Float16 T[32][33];
    const int e  = blockIdx.z;
    const int n0 = blockIdx.x * 32, k0 = blockIdx.y * 32;
    const float*   src = W  + ((size_t)e * K + k0) * N + n0;
    _Float16*      dst = Wt + ((size_t)e * N + n0) * K + k0;
    const int tx = threadIdx.x & 31, ty = threadIdx.x >> 5;   // 8 rows / pass
#pragma unroll
    for (int p = 0; p < 4; ++p) {
        int k = ty + p * 8;
        T[tx][k] = (_Float16)src[(size_t)k * N + tx];
    }
    __syncthreads();
#pragma unroll
    for (int p = 0; p < 2; ++p) {
        int idx = p * 256 + threadIdx.x;          // 512 half2 units
        int n   = idx >> 4;
        int c2  = (idx & 15) * 2;
        half2_t h = { T[n][c2], T[n][c2 + 1] };
        *(half2_t*)(dst + (size_t)n * K + c2) = h;
    }
}

// ---------------- gating: logits -> softmax -> top2 -> renorm -> route ----------------
__global__ void __launch_bounds__(256) gate_kernel(const float* __restrict__ x,
                                                   const float* __restrict__ Wg,
                                                   const float* __restrict__ bg,
                                                   int* __restrict__ cnt,
                                                   int* __restrict__ tok,
                                                   float* __restrict__ gatew) {
    const int t    = blockIdx.x * 8 + (threadIdx.x >> 5);
    const int lane = threadIdx.x & 31;
    float l[E];
#pragma unroll
    for (int e = 0; e < E; ++e) l[e] = 0.f;
    for (int d = lane; d < I; d += 32) {
        float xv = x[(size_t)t * I + d];
        float4 w0 = *(const float4*)(Wg + (size_t)d * E);
        float4 w1 = *(const float4*)(Wg + (size_t)d * E + 4);
        l[0] += xv * w0.x; l[1] += xv * w0.y; l[2] += xv * w0.z; l[3] += xv * w0.w;
        l[4] += xv * w1.x; l[5] += xv * w1.y; l[6] += xv * w1.z; l[7] += xv * w1.w;
    }
#pragma unroll
    for (int off = 16; off > 0; off >>= 1)
#pragma unroll
        for (int e = 0; e < E; ++e) l[e] += __shfl_down(l[e], off, 32);
    if (lane == 0) {
        float mx = -1e30f;
#pragma unroll
        for (int e = 0; e < E; ++e) { l[e] += bg[e]; mx = fmaxf(mx, l[e]); }
        float p[E];
#pragma unroll
        for (int e = 0; e < E; ++e) p[e] = __expf(l[e] - mx);
        int e0 = 0; float v0 = p[0];
#pragma unroll
        for (int e = 1; e < E; ++e) if (p[e] > v0) { v0 = p[e]; e0 = e; }
        int e1 = -1; float v1 = -1.f;
#pragma unroll
        for (int e = 0; e < E; ++e) if (e != e0 && p[e] > v1) { v1 = p[e]; e1 = e; }
        float inv = 1.f / (v0 + v1);
        int r0 = atomicAdd(&cnt[e0], 1);
        tok[e0 * B + r0] = t;  gatew[e0 * B + r0] = v0 * inv;
        int r1 = atomicAdd(&cnt[e1], 1);
        tok[e1 * B + r1] = t;  gatew[e1 * B + r1] = v1 * inv;
    }
}

__global__ void offsets_kernel(const int* __restrict__ cnt, int* __restrict__ offs) {
    if (threadIdx.x == 0) {
        int s = 0;
#pragma unroll
        for (int e = 0; e < E; ++e) { offs[e] = s; s += cnt[e]; }
        offs[E] = s;
    }
}

// ---------------- GEMM1: h[slot] = relu(x[tok[slot]] @ W1[e] + b1[e]) ----------------
// A: per-lane async gather (f16 x).  B: TDM tile of N-major W1t.  Double-buffered LDS.
__global__ void __launch_bounds__(256) gemm1_kernel(const _Float16* __restrict__ xh,
                                                    const _Float16* __restrict__ W1t,
                                                    const float* __restrict__ b1,
                                                    const int* __restrict__ cnt,
                                                    const int* __restrict__ offs,
                                                    const int* __restrict__ tok,
                                                    _Float16* __restrict__ hbuf) {
    __shared__ __align__(16) _Float16 As[2][BM * LDA];
    __shared__ __align__(16) _Float16 Bs[2][BN * LDB];
    __shared__ int tokS[BM];

    const int e  = blockIdx.z;
    const int cn = cnt[e];
    const int m0 = blockIdx.y * BM;
    if (m0 >= cn) return;
    const int nb  = blockIdx.x * BN;
    const int off = offs[e];
    const int tid = threadIdx.x, lane = tid & 31, wave = tid >> 5;
    const int wm = wave >> 2, wn = wave & 3;
    const _Float16* WtE = W1t + ((size_t)e * H + nb) * I;   // rows nb.., K = I

    if (tid < BM) tokS[tid] = (m0 + tid < cn) ? tok[e * B + m0 + tid] : 0;
    __syncthreads();

    // A tile = 512 x 16B granules; 2 per thread.  granule g: row g>>2, col (g&3)*8 halves
    const int g0i = tid * 2,     r0 = g0i >> 2, c0h = (g0i & 3) * 8;
    const int g1i = tid * 2 + 1, r1 = g1i >> 2, c1h = (g1i & 3) * 8;
    const _Float16* aSrc0 = xh + (size_t)tokS[r0] * I + c0h;
    const _Float16* aSrc1 = xh + (size_t)tokS[r1] * I + c1h;
    const uint32_t asBase = (uint32_t)(uintptr_t)&As[0][0];
    const uint32_t bsBase = (uint32_t)(uintptr_t)&Bs[0][0];
    const uint32_t aDst0 = asBase + (uint32_t)(r0 * LDA + c0h) * 2;
    const uint32_t aDst1 = asBase + (uint32_t)(r1 * LDA + c1h) * 2;
    constexpr uint32_t ABUF = BM * LDA * 2;
    constexpr uint32_t BBUF = BN * LDB * 2;

    // prologue: stage kt=0 into buffer 0
    async_b128(aDst0, aSrc0);
    async_b128(aDst1, aSrc1);
    if (wave == 0) tdm_load_2d(bsBase, WtE, BK, BN, BK, BN, (uint64_t)I);
    wait_async0();
    if (wave == 0) __builtin_amdgcn_s_wait_tensorcnt(0);
    __syncthreads();

    float8_t acc[4][4] = {};
    int cur = 0;
    for (int kt = 0; kt < I; kt += BK) {
        const int  nxt     = cur ^ 1;
        const bool hasNext = (kt + BK) < I;
        if (hasNext) {
            const int kn = kt + BK;
            async_b128(aDst0 + nxt * ABUF, aSrc0 + kn);
            async_b128(aDst1 + nxt * ABUF, aSrc1 + kn);
            if (wave == 0) tdm_load_2d(bsBase + nxt * BBUF, WtE + kn, BK, BN, BK, BN, (uint64_t)I);
        }
        const _Float16* Ab = &As[cur][0];
        const _Float16* Bb = &Bs[cur][0];
        half16_t af[4], bf[4];
#pragma unroll
        for (int fm = 0; fm < 4; ++fm) af[fm] = frag_a(Ab, wm * 64 + fm * 16, lane);
#pragma unroll
        for (int fn = 0; fn < 4; ++fn) bf[fn] = frag_b(Bb, wn * 64 + fn * 16, lane);
#pragma unroll
        for (int fm = 0; fm < 4; ++fm)
#pragma unroll
            for (int fn = 0; fn < 4; ++fn)
                acc[fm][fn] = __builtin_amdgcn_wmma_f32_16x16x32_f16(
                    false, af[fm], false, bf[fn], (short)0, acc[fm][fn], false, false);
        if (hasNext) {
            wait_async0();
            if (wave == 0) __builtin_amdgcn_s_wait_tensorcnt(0);
        }
        __syncthreads();
        cur = nxt;
    }

    const float* b1e = b1 + (size_t)e * H;
#pragma unroll
    for (int fn = 0; fn < 4; ++fn) {
        int ncol = nb + wn * 64 + fn * 16 + (lane & 15);
        float bv = b1e[ncol];
#pragma unroll
        for (int fm = 0; fm < 4; ++fm) {
            int mrel = wm * 64 + fm * 16 + ((lane >> 4) << 3);
#pragma unroll
            for (int r = 0; r < 8; ++r) {
                int mm = m0 + mrel + r;
                if (mm < cn) {
                    float v = acc[fm][fn][r] + bv;
                    hbuf[(size_t)(off + mm) * H + ncol] = (_Float16)fmaxf(v, 0.f);
                }
            }
        }
    }
}

// ---------------- GEMM2: out[tok] += gate * (h @ W2[e] + b2[e]) ----------------
// A and B tiles both via TDM (h already f16, row-clamped by tensor_dim1).
__global__ void __launch_bounds__(256) gemm2_kernel(const _Float16* __restrict__ hbuf,
                                                    const _Float16* __restrict__ W2t,
                                                    const float* __restrict__ b2,
                                                    const int* __restrict__ cnt,
                                                    const int* __restrict__ offs,
                                                    const int* __restrict__ tok,
                                                    const float* __restrict__ gatew,
                                                    float* __restrict__ out) {
    __shared__ __align__(16) _Float16 As[2][BM * LDA];
    __shared__ __align__(16) _Float16 Bs[2][BN * LDB];

    const int e  = blockIdx.z;
    const int cn = cnt[e];
    const int m0 = blockIdx.y * BM;
    if (m0 >= cn) return;
    const int nb  = blockIdx.x * BN;
    const int off = offs[e];
    const int tid = threadIdx.x, lane = tid & 31, wave = tid >> 5;
    const int wm = wave >> 2, wn = wave & 3;
    const _Float16* WtE  = W2t + ((size_t)e * O + nb) * H;        // rows nb.., K = H
    const _Float16* Arow = hbuf + (size_t)(off + m0) * H;         // tile row 0
    const uint32_t rows = (uint32_t)(cn - m0);

    const uint32_t asBase = (uint32_t)(uintptr_t)&As[0][0];
    const uint32_t bsBase = (uint32_t)(uintptr_t)&Bs[0][0];
    constexpr uint32_t ABUF = BM * LDA * 2;
    constexpr uint32_t BBUF = BN * LDB * 2;

    if (wave == 0) {
        tdm_load_2d(asBase, Arow, BK, rows, BK, BM, (uint64_t)H);
        tdm_load_2d(bsBase, WtE,  BK, BN,   BK, BN, (uint64_t)H);
        __builtin_amdgcn_s_wait_tensorcnt(0);
    }
    __syncthreads();

    float8_t acc[4][4] = {};
    int cur = 0;
    for (int kt = 0; kt < H; kt += BK) {
        const int  nxt     = cur ^ 1;
        const bool hasNext = (kt + BK) < H;
        if (hasNext && wave == 0) {
            const int kn = kt + BK;
            tdm_load_2d(asBase + nxt * ABUF, Arow + kn, BK, rows, BK, BM, (uint64_t)H);
            tdm_load_2d(bsBase + nxt * BBUF, WtE  + kn, BK, BN,   BK, BN, (uint64_t)H);
        }
        const _Float16* Ab = &As[cur][0];
        const _Float16* Bb = &Bs[cur][0];
        half16_t af[4], bf[4];
#pragma unroll
        for (int fm = 0; fm < 4; ++fm) af[fm] = frag_a(Ab, wm * 64 + fm * 16, lane);
#pragma unroll
        for (int fn = 0; fn < 4; ++fn) bf[fn] = frag_b(Bb, wn * 64 + fn * 16, lane);
#pragma unroll
        for (int fm = 0; fm < 4; ++fm)
#pragma unroll
            for (int fn = 0; fn < 4; ++fn)
                acc[fm][fn] = __builtin_amdgcn_wmma_f32_16x16x32_f16(
                    false, af[fm], false, bf[fn], (short)0, acc[fm][fn], false, false);
        if (hasNext && wave == 0) __builtin_amdgcn_s_wait_tensorcnt(0);
        __syncthreads();
        cur = nxt;
    }

    const float* b2e = b2 + (size_t)e * O;
#pragma unroll
    for (int fn = 0; fn < 4; ++fn) {
        int ncol = nb + wn * 64 + fn * 16 + (lane & 15);
        float bv = b2e[ncol];
#pragma unroll
        for (int fm = 0; fm < 4; ++fm) {
            int mrel = wm * 64 + fm * 16 + ((lane >> 4) << 3);
#pragma unroll
            for (int r = 0; r < 8; ++r) {
                int mm = m0 + mrel + r;
                if (mm < cn) {
                    int   t = tok[e * B + mm];
                    float g = gatew[e * B + mm];
                    atomicAdd(out + (size_t)t * O + ncol, g * (acc[fm][fn][r] + bv));
                }
            }
        }
    }
}

// ---------------- host launch ----------------
extern "C" void kernel_launch(void* const* d_in, const int* in_sizes, int n_in,
                              void* d_out, int out_size, void* d_ws, size_t ws_size,
                              hipStream_t stream) {
    (void)in_sizes; (void)n_in; (void)out_size; (void)ws_size;
    const float* x  = (const float*)d_in[0];
    const float* Wg = (const float*)d_in[1];
    const float* bg = (const float*)d_in[2];
    const float* W1 = (const float*)d_in[3];
    const float* b1 = (const float*)d_in[4];
    const float* W2 = (const float*)d_in[5];
    const float* b2 = (const float*)d_in[6];
    float* out = (float*)d_out;

    char* ws = (char*)d_ws;
    int*      cnt   = (int*)(ws);                          // 64 B
    int*      offs  = (int*)(ws + 64);                     // 64 B
    int*      tok   = (int*)(ws + 128);                    // 32 KB
    float*    gatew = (float*)(ws + 128 + E * B * 4);      // 32 KB
    _Float16* xh    = (_Float16*)(ws + 65792);             // 1 MB
    _Float16* W1t   = (_Float16*)(ws + 65792 + 1048576);   // 16 MB  [E][H][I]
    _Float16* W2t   = (_Float16*)(ws + 65792 + 1048576 + 16777216);   // 16 MB [E][O][H]
    _Float16* hbuf  = (_Float16*)(ws + 65792 + 1048576 + 2 * 16777216); // 8 MB

    hipMemsetAsync(cnt, 0, 64, stream);
    hipMemsetAsync(out, 0, (size_t)B * O * sizeof(float), stream);

    cvt_x_kernel<<<(B * I / 4) / 256, 256, 0, stream>>>(x, xh);
    transpose_cvt_kernel<<<dim3(H / 32, I / 32, E), 256, 0, stream>>>(W1, W1t, I, H);
    transpose_cvt_kernel<<<dim3(O / 32, H / 32, E), 256, 0, stream>>>(W2, W2t, H, O);
    gate_kernel<<<B / 8, 256, 0, stream>>>(x, Wg, bg, cnt, tok, gatew);
    offsets_kernel<<<1, 32, 0, stream>>>(cnt, offs);
    gemm1_kernel<<<dim3(H / BN, B / BM, E), 256, 0, stream>>>(xh, W1t, b1, cnt, offs, tok, hbuf);
    gemm2_kernel<<<dim3(O / BN, B / BM, E), 256, 0, stream>>>(hbuf, W2t, b2, cnt, offs, tok, gatew, out);
}
